// TiedGateIntegrator_12730283066055
// MI455X (gfx1250) — matrix operand order repaired and checked by reading, not compiled, not run on hardware
//
#include <hip/hip_runtime.h>
#include <stdint.h>

// ---------------------------------------------------------------------------
// Types for CDNA5 WMMA
// ---------------------------------------------------------------------------
typedef __bf16 v16bf __attribute__((ext_vector_type(16)));
typedef float  v8f   __attribute__((ext_vector_type(8)));

union FragU {
    v16bf v;
    uint4 q[2];
};

__device__ __forceinline__ unsigned short f2bf(float f) {
    unsigned int u = __float_as_uint(f);
    u += 0x7FFFu + ((u >> 16) & 1u);   // round-to-nearest-even to bf16
    return (unsigned short)(u >> 16);
}

// ---------------------------------------------------------------------------
// Problem constants (fixed by the reference)
// ---------------------------------------------------------------------------
static constexpr int BB = 4;
static constexpr int SS = 4096;
static constexpr int DD = 1024;
static constexpr int MM = BB * SS;       // 16384 rows
static constexpr int CHUNK = 256;        // scan chunk length
static constexpr int NCH  = SS / CHUNK;  // 16 chunks per (b,d) lane

// ---------------------------------------------------------------------------
// Prep: fp32 -> bf16 (vectorized), and weight transpose+convert
// ---------------------------------------------------------------------------
__global__ void cvt_f32_to_bf16_x4(const float4* __restrict__ in,
                                   ushort4* __restrict__ out) {
    int i = blockIdx.x * blockDim.x + threadIdx.x;
    float4 f = in[i];
    ushort4 u;
    u.x = f2bf(f.x); u.y = f2bf(f.y); u.z = f2bf(f.z); u.w = f2bf(f.w);
    out[i] = u;
}

// Wt[n][k] = bf16(W[k][n]); W is DD x DD row-major.
__global__ void transpose_cvt(const float* __restrict__ W,
                              unsigned short* __restrict__ Wt) {
    __shared__ float tile[32][33];
    const int n0 = blockIdx.x * 32;
    const int k0 = blockIdx.y * 32;
    const int tx = threadIdx.x, ty = threadIdx.y;
#pragma unroll
    for (int i = 0; i < 4; ++i)
        tile[ty + i * 8][tx] = W[(size_t)(k0 + ty + i * 8) * DD + n0 + tx];
    __syncthreads();
#pragma unroll
    for (int i = 0; i < 4; ++i)
        Wt[(size_t)(n0 + ty + i * 8) * DD + k0 + tx] = f2bf(tile[tx][ty + i * 8]);
}

// ---------------------------------------------------------------------------
// bf16 WMMA GEMM:  out[M,N] = act(A[M,K] * Wt[N,K]^T + bias)
//   mode 0: relu  -> bf16 outH                       (g1)
//   mode 1: 1/(1+e^z) -> fp32 outF                   (a = 1 - sigmoid(z))
//   mode 2: z*(1/32); outF = gate * that -> fp32     (b_gated = a*b)
// Block tile 128x128, 256 threads = 8 waves (4x2 grid of 32x64 wave tiles),
// each wave: 2x4 of 16x16 tiles via V_WMMA_F32_16X16X32_BF16, K-step 32.
// ---------------------------------------------------------------------------
__global__ __launch_bounds__(256)
void gemm_bf16_wmma(const unsigned short* __restrict__ A,
                    const unsigned short* __restrict__ Bt,
                    const float* __restrict__ bias,
                    const float* __restrict__ gate,
                    float* outF,
                    unsigned short* outH,
                    int mode) {
    constexpr int K   = DD;
    constexpr int N   = DD;
    constexpr int LDT = 40;  // padded LDS row stride (elements); 80B, 16B-aligned

    __shared__ __align__(16) unsigned short As[128 * LDT];
    __shared__ __align__(16) unsigned short Bs[128 * LDT];

    const int m0   = blockIdx.y * 128;
    const int n0   = blockIdx.x * 128;
    const int tid  = threadIdx.x;
    const int lane = tid & 31;
    const int wid  = tid >> 5;
    const int wm   = wid >> 1;       // 0..3 : 32-row band
    const int wn   = wid & 1;        // 0..1 : 64-col band
    const int l16  = lane & 15;
    const int lh   = lane >> 4;      // half-wave select

    v8f acc[2][4];
#pragma unroll
    for (int mi = 0; mi < 2; ++mi)
#pragma unroll
        for (int ni = 0; ni < 4; ++ni)
            acc[mi][ni] = (v8f)0.0f;

    for (int kb = 0; kb < K; kb += 32) {
        // Fill 128x32 bf16 tiles: 512 16B chunks per tile, 2 per thread.
#pragma unroll
        for (int i = 0; i < 2; ++i) {
            int idx = tid + i * 256;          // 0..511
            int row = idx >> 2;               // 0..127
            int c8  = (idx & 3) * 8;          // 0,8,16,24 elements
            uint4 va = *reinterpret_cast<const uint4*>(
                &A[(size_t)(m0 + row) * K + kb + c8]);
            *reinterpret_cast<uint4*>(&As[row * LDT + c8]) = va;
            uint4 vb = *reinterpret_cast<const uint4*>(
                &Bt[(size_t)(n0 + row) * K + kb + c8]);
            *reinterpret_cast<uint4*>(&Bs[row * LDT + c8]) = vb;
        }
        __syncthreads();

        // A fragments (16x32, ISA 7.12.2): lane half lh holds K runs
        // {lh*8..+7} and {16+lh*8..+7} for row l16 of the tile.
        FragU fa[2];
#pragma unroll
        for (int mi = 0; mi < 2; ++mi) {
            int r = (wm * 32 + mi * 16 + l16) * LDT;
            fa[mi].q[0] = *reinterpret_cast<const uint4*>(&As[r + lh * 8]);
            fa[mi].q[1] = *reinterpret_cast<const uint4*>(&As[r + 16 + lh * 8]);
        }
        // B fragments (32x16): lane half lh holds K runs
        // {lh*8..+7} and {16+lh*8..+7} for column l16 (matches A).
        FragU fb[4];
#pragma unroll
        for (int ni = 0; ni < 4; ++ni) {
            int c = (wn * 64 + ni * 16 + l16) * LDT;
            fb[ni].q[0] = *reinterpret_cast<const uint4*>(&Bs[c + lh * 8]);
            fb[ni].q[1] = *reinterpret_cast<const uint4*>(&Bs[c + 16 + lh * 8]);
        }

#pragma unroll
        for (int mi = 0; mi < 2; ++mi)
#pragma unroll
            for (int ni = 0; ni < 4; ++ni)
                acc[mi][ni] = __builtin_amdgcn_wmma_f32_16x16x32_bf16(
                    false, fa[mi].v, false, fb[ni].v,
                    (short)0, acc[mi][ni], false, false);
        __syncthreads();
    }

    // Epilogue. C/D layout: VGPR r holds M = r + 8*lh, N = l16 (per 16x16 tile).
#pragma unroll
    for (int mi = 0; mi < 2; ++mi) {
#pragma unroll
        for (int ni = 0; ni < 4; ++ni) {
            const int gcol = n0 + wn * 64 + ni * 16 + l16;
            const float bv = bias[gcol];
#pragma unroll
            for (int r = 0; r < 8; ++r) {
                const int grow = m0 + wm * 32 + mi * 16 + lh * 8 + r;
                const size_t o = (size_t)grow * N + gcol;
                float v = acc[mi][ni][r] + bv;
                if (mode == 0) {
                    outH[o] = f2bf(v > 0.0f ? v : 0.0f);
                } else if (mode == 1) {
                    outF[o] = 1.0f / (1.0f + __expf(v));   // 1 - sigmoid(v)
                } else {
                    outF[o] = gate[o] * (v * 0.03125f);    // a * b/sqrt(D)
                }
            }
        }
    }
}

// ---------------------------------------------------------------------------
// Chunked scan of h <- a*h + bg along S, per (b,d) lane.
// ---------------------------------------------------------------------------
__global__ void scan_chunks(const float* __restrict__ a,
                            const float* __restrict__ bg,
                            float* __restrict__ Aagg,
                            float* __restrict__ Bagg) {
    int tid = blockIdx.x * blockDim.x + threadIdx.x;  // (b*NCH + c)*DD + d
    int d  = tid & (DD - 1);
    int bc = tid >> 10;
    int c  = bc & (NCH - 1);
    int b  = bc >> 4;
    size_t base = ((size_t)b * SS + (size_t)c * CHUNK) * DD + d;
    float Av = 1.0f, Bv = 0.0f;
#pragma unroll 4
    for (int t = 0; t < CHUNK; ++t) {
        float av = a[base + (size_t)t * DD];
        float bv = bg[base + (size_t)t * DD];
        Bv = fmaf(av, Bv, bv);
        Av *= av;
    }
    Aagg[tid] = Av;
    Bagg[tid] = Bv;
}

__global__ void scan_carries(const float* __restrict__ Aagg,
                             const float* __restrict__ Bagg,
                             float* __restrict__ carry) {
    int tid = blockIdx.x * blockDim.x + threadIdx.x;  // b*DD + d
    int d = tid & (DD - 1);
    int b = tid >> 10;
    float h = 0.0f;
#pragma unroll
    for (int c = 0; c < NCH; ++c) {
        int i = (b * NCH + c) * DD + d;
        carry[i] = h;                      // exclusive carry into chunk c
        h = fmaf(Aagg[i], h, Bagg[i]);
    }
}

// a and out may alias (a lives in d_out); each slot is read before written
// by the same thread, so in-place is safe.
__global__ void scan_apply(const float* a,
                           const float* __restrict__ bg,
                           const float* __restrict__ carry,
                           float* out) {
    int tid = blockIdx.x * blockDim.x + threadIdx.x;
    int d  = tid & (DD - 1);
    int bc = tid >> 10;
    int c  = bc & (NCH - 1);
    int b  = bc >> 4;
    size_t base = ((size_t)b * SS + (size_t)c * CHUNK) * DD + d;
    float h = carry[tid];
#pragma unroll 4
    for (int t = 0; t < CHUNK; ++t) {
        size_t i = base + (size_t)t * DD;
        float av = a[i];
        float bv = bg[i];
        h = fmaf(av, h, bv);
        out[i] = h;
    }
}

// ---------------------------------------------------------------------------
// Launch
// ---------------------------------------------------------------------------
extern "C" void kernel_launch(void* const* d_in, const int* in_sizes, int n_in,
                              void* d_out, int out_size, void* d_ws, size_t ws_size,
                              hipStream_t stream) {
    const float* x   = (const float*)d_in[0];
    const float* Wg1 = (const float*)d_in[1];
    const float* bg1 = (const float*)d_in[2];
    const float* Wg2 = (const float*)d_in[3];
    const float* bg2 = (const float*)d_in[4];
    const float* Wdr = (const float*)d_in[5];
    const float* bdr = (const float*)d_in[6];
    float* out = (float*)d_out;  // holds `a` after GEMM2, final states at end

    // Workspace carve-up (all offsets 256B-aligned)
    char* ws = (char*)d_ws;
    size_t off = 0;
    unsigned short* xbf  = (unsigned short*)(ws + off); off += (size_t)MM * DD * 2;
    unsigned short* g1bf = (unsigned short*)(ws + off); off += (size_t)MM * DD * 2;
    unsigned short* wt1  = (unsigned short*)(ws + off); off += (size_t)DD * DD * 2;
    unsigned short* wt2  = (unsigned short*)(ws + off); off += (size_t)DD * DD * 2;
    unsigned short* wt3  = (unsigned short*)(ws + off); off += (size_t)DD * DD * 2;
    float* bgbuf = (float*)(ws + off); off += (size_t)MM * DD * 4;
    float* Aagg  = (float*)(ws + off); off += (size_t)BB * NCH * DD * 4;
    float* Bagg  = (float*)(ws + off); off += (size_t)BB * NCH * DD * 4;
    float* carry = (float*)(ws + off); off += (size_t)BB * NCH * DD * 4;

    // 1) x -> bf16
    cvt_f32_to_bf16_x4<<<(MM * DD) / (256 * 4), 256, 0, stream>>>(
        (const float4*)x, (ushort4*)xbf);

    // 2) weights -> bf16, transposed to [N][K]
    dim3 tb(32, 8), tg(DD / 32, DD / 32);
    transpose_cvt<<<tg, tb, 0, stream>>>(Wg1, wt1);
    transpose_cvt<<<tg, tb, 0, stream>>>(Wg2, wt2);
    transpose_cvt<<<tg, tb, 0, stream>>>(Wdr, wt3);

    // 3) g1 = relu(x @ Wg1 + b1)                (bf16)
    // 4) a  = 1 - sigmoid(g1 @ Wg2 + b2)        (fp32, into d_out)
    // 5) bg = a * (x @ Wdr + b3)/32             (fp32)
    dim3 gg(DD / 128, MM / 128), gb(256);
    gemm_bf16_wmma<<<gg, gb, 0, stream>>>(xbf,  wt1, bg1, nullptr, nullptr, g1bf, 0);
    gemm_bf16_wmma<<<gg, gb, 0, stream>>>(g1bf, wt2, bg2, nullptr, out,     nullptr, 1);
    gemm_bf16_wmma<<<gg, gb, 0, stream>>>(xbf,  wt3, bdr, out,     bgbuf,   nullptr, 2);

    // 6-8) chunked linear-recurrence scan along S
    scan_chunks <<<(BB * NCH * DD) / 256, 256, 0, stream>>>(out, bgbuf, Aagg, Bagg);
    scan_carries<<<(BB * DD) / 256,       256, 0, stream>>>(Aagg, Bagg, carry);
    scan_apply  <<<(BB * NCH * DD) / 256, 256, 0, stream>>>(out, bgbuf, carry, out);

    (void)in_sizes; (void)n_in; (void)out_size; (void)ws_size;
}
